// LearnableUpsample_69827578298482
// MI455X (gfx1250) — compile-verified, matrix operand order checked by reading, not run on hardware
//
#include <hip/hip_runtime.h>
#include <hip/hip_bf16.h>

typedef __bf16 v16bf __attribute__((ext_vector_type(16)));
typedef __bf16 bf4   __attribute__((ext_vector_type(4)));
typedef float  v8f   __attribute__((ext_vector_type(8)));
typedef int    v4i   __attribute__((ext_vector_type(4)));

constexpr int E    = 600000;
constexpr int NF   = 50000;
constexpr int D    = 128;
constexpr int HID  = 128;
constexpr int DIN  = 512;   // 4*D

constexpr int BE   = 64;    // edges per block (E % BE == 0 -> 9375 blocks)
constexpr int TPB  = 256;   // 8 wave32
constexpr int XPITCH = DIN + 16;  // bf16 elements, row stride 1056B (32B aligned, bank-spread)
constexpr int HPITCH = HID + 16;  // 288B rows

#define HAS_ASYNC_LDS __has_builtin(__builtin_amdgcn_global_load_async_to_lds_b128)

#if HAS_ASYNC_LDS
__device__ __forceinline__ void async_copy16(const void* g, void* l) {
    typedef __attribute__((address_space(1))) v4i gv4i_t;   // global int4
    typedef __attribute__((address_space(3))) v4i lv4i_t;   // LDS int4
    __builtin_amdgcn_global_load_async_to_lds_b128(
        (gv4i_t*)(v4i*)const_cast<void*>(g), (lv4i_t*)(v4i*)l, 0, 0);
}
__device__ __forceinline__ void wait_async0() {
#if __has_builtin(__builtin_amdgcn_s_wait_asynccnt)
    __builtin_amdgcn_s_wait_asynccnt(0);
#else
    asm volatile("s_wait_asynccnt 0" ::: "memory");
#endif
}
#endif

// ---------------------------------------------------------------------------
// Prep: transpose + convert weights to bf16 in workspace.
// w1t: [HID][DIN] bf16 (row n holds W1[:,n]) ; w2t: [D][HID] bf16 (row n holds W2[:,n])
// ---------------------------------------------------------------------------
__global__ __launch_bounds__(TPB) void prep_weights(const float* __restrict__ W1,
                                                    const float* __restrict__ W2,
                                                    __bf16* __restrict__ w1t,
                                                    __bf16* __restrict__ w2t) {
    int i = blockIdx.x * TPB + threadIdx.x;
    if (i < DIN * HID) {
        int k = i / HID, n = i % HID;
        w1t[n * DIN + k] = (__bf16)W1[i];
    }
    if (i < HID * D) {
        int k = i / D, n = i % D;
        w2t[n * HID + k] = (__bf16)W2[i];
    }
}

// ---------------------------------------------------------------------------
// Fused: gather -> x(bf16) -> WMMA GEMM1 -> relu -> WMMA GEMM2 -> relu
//        -> *h_dk -> atomic scatter-add
// ---------------------------------------------------------------------------
__global__ __launch_bounds__(TPB) void edge_mlp_scatter(
    const float* __restrict__ h_dk, const float* __restrict__ h_si,
    const float* __restrict__ h_sk, const float* __restrict__ h_di,
    const int* __restrict__ fine_idx, const int* __restrict__ coarse_idx,
    const __bf16* __restrict__ w1t, const __bf16* __restrict__ w2t,
    const float* __restrict__ b1, const float* __restrict__ b2,
    float* __restrict__ out) {

    __shared__ __align__(32) __bf16 w1s[HID * XPITCH];  // 135168 B
    __shared__ __align__(32) __bf16 xs [BE  * XPITCH];  //  67584 B
    __shared__ __align__(32) __bf16 w2s[HID * HPITCH];  //  36864 B
    __shared__ __align__(32) __bf16 hs [BE  * HPITCH];  //  18432 B
    __shared__ int fidxs[BE];
    __shared__ int cidxs[BE];

    const int tid  = threadIdx.x;
    const int lane = tid & 31;
    const int wv   = tid >> 5;
    const int e0   = blockIdx.x * BE;

    // --- stage per-edge indices ---
    if (tid < BE) {
        fidxs[tid] = fine_idx[e0 + tid];
        cidxs[tid] = coarse_idx[e0 + tid];
    }

    // --- stage W1^T / W2^T into LDS (16B chunks) ---
    // Async path: VGPR-free global->LDS DMA, overlapped with the gather phase.
#if HAS_ASYNC_LDS
    {
        const uint4* s1 = (const uint4*)w1t;                 // 8 bf16 per chunk
        for (int i = tid; i < HID * (DIN / 8); i += TPB) {
            int r = i >> 6, c = i & 63;
            async_copy16(&s1[r * 64 + c], &w1s[r * XPITCH + c * 8]);
        }
        const uint4* s2 = (const uint4*)w2t;
        for (int i = tid; i < HID * (HID / 8); i += TPB) {
            int r = i >> 4, c = i & 15;
            async_copy16(&s2[r * 16 + c], &w2s[r * HPITCH + c * 8]);
        }
    }
#else
    {
        const uint4* s1 = (const uint4*)w1t;                 // 8 bf16 per chunk
        for (int i = tid; i < HID * (DIN / 8); i += TPB) {
            int r = i >> 6, c = i & 63;
            *(uint4*)&w1s[r * XPITCH + c * 8] = s1[r * 64 + c];
        }
        const uint4* s2 = (const uint4*)w2t;
        for (int i = tid; i < HID * (HID / 8); i += TPB) {
            int r = i >> 4, c = i & 15;
            *(uint4*)&w2s[r * HPITCH + c * 8] = s2[r * 16 + c];
        }
    }
#endif
    __syncthreads();   // indices (and, on fallback, weights) visible

    // --- gather & convert x = [h_si | h_sk | h_di | h_dk] -> bf16 in LDS ---
    {
        const int e   = tid >> 2;        // 0..63
        const int blk = tid & 3;         // which 128-wide block
        const int fi = fidxs[e], ci = cidxs[e];
        const float* src = (blk == 0) ? (h_si + (size_t)fi * D)
                         : (blk == 1) ? (h_sk + (size_t)ci * D)
                         : (blk == 2) ? (h_di + (size_t)fi * D)
                                      : (h_dk + (size_t)ci * D);
        __bf16* dst = &xs[e * XPITCH + blk * D];
        #pragma unroll
        for (int j = 0; j < D / 4; ++j) {
            float4 v = ((const float4*)src)[j];
            bf4 p = { (__bf16)v.x, (__bf16)v.y, (__bf16)v.z, (__bf16)v.w };
            *(bf4*)&dst[j * 4] = p;      // 8B LDS store
        }
    }
#if HAS_ASYNC_LDS
    wait_async0();                       // weight DMA complete (per-wave)
#endif
    __syncthreads();                     // xs + w1s/w2s visible to all waves

    // wave -> tile mapping: 4 Mtiles x 8 Ntiles, each wave owns (mt, 4 ntiles)
    const int mt = wv >> 1;          // 0..3  (16 edges each)
    const int n0 = (wv & 1) * 4;     // ntile base 0 or 4
    const int lrow = lane & 15;      // row/col within fragment
    const int khalf = (lane >> 4) * 16;

    // ---------------- GEMM1: [64 x 512] @ [512 x 128] ----------------
    v8f acc[4] = {};
    for (int kt = 0; kt < DIN / 32; ++kt) {
        const int kof = kt * 32 + khalf;
        v16bf a = *(const v16bf*)&xs[(mt * 16 + lrow) * XPITCH + kof];
        #pragma unroll
        for (int t = 0; t < 4; ++t) {
            v16bf b = *(const v16bf*)&w1s[((n0 + t) * 16 + lrow) * XPITCH + kof];
            acc[t] = __builtin_amdgcn_wmma_f32_16x16x32_bf16(
                false, a, false, b, (short)0, acc[t], false, false);
        }
    }

    // bias + relu -> bf16 h in LDS
    #pragma unroll
    for (int t = 0; t < 4; ++t) {
        const int n = (n0 + t) * 16 + lrow;
        const float bb = b1[n];
        #pragma unroll
        for (int j = 0; j < 8; ++j) {
            const int m = mt * 16 + j + ((lane >> 4) << 3);
            float v = fmaxf(acc[t][j] + bb, 0.0f);
            hs[m * HPITCH + n] = (__bf16)v;
        }
    }
    __syncthreads();

    // ---------------- GEMM2: [64 x 128] @ [128 x 128] ----------------
    v8f acc2[4] = {};
    for (int kt = 0; kt < HID / 32; ++kt) {
        const int kof = kt * 32 + khalf;
        v16bf a = *(const v16bf*)&hs[(mt * 16 + lrow) * HPITCH + kof];
        #pragma unroll
        for (int t = 0; t < 4; ++t) {
            v16bf b = *(const v16bf*)&w2s[((n0 + t) * 16 + lrow) * HPITCH + kof];
            acc2[t] = __builtin_amdgcn_wmma_f32_16x16x32_bf16(
                false, a, false, b, (short)0, acc2[t], false, false);
        }
    }

    // ---- epilogue: relu(psi + b2) * h_dk_g, scatter-add to fine nodes ----
    #pragma unroll
    for (int t = 0; t < 4; ++t) {
        const int n = (n0 + t) * 16 + lrow;
        const float bb = b2[n];
        #pragma unroll
        for (int j = 0; j < 8; ++j) {
            const int m = mt * 16 + j + ((lane >> 4) << 3);
            const float psi = fmaxf(acc2[t][j] + bb, 0.0f);
            const float contrib = psi * h_dk[(size_t)cidxs[m] * D + n];
            atomicAdd(&out[(size_t)fidxs[m] * D + n], contrib);
        }
    }
}

// ---------------------------------------------------------------------------
extern "C" void kernel_launch(void* const* d_in, const int* in_sizes, int n_in,
                              void* d_out, int out_size, void* d_ws, size_t ws_size,
                              hipStream_t stream) {
    const float* h_dk = (const float*)d_in[0];
    const float* h_si = (const float*)d_in[1];
    const float* h_sk = (const float*)d_in[2];
    const float* h_di = (const float*)d_in[3];
    const int*   fidx = (const int*)d_in[4];
    const int*   cidx = (const int*)d_in[5];
    const float* W1   = (const float*)d_in[6];
    const float* b1   = (const float*)d_in[7];
    const float* W2   = (const float*)d_in[8];
    const float* b2   = (const float*)d_in[9];
    float* out = (float*)d_out;

    __bf16* w1t = (__bf16*)d_ws;
    __bf16* w2t = (__bf16*)((char*)d_ws + (size_t)DIN * HID * sizeof(__bf16));

    (void)hipMemsetAsync(d_out, 0, (size_t)NF * D * sizeof(float), stream);
    prep_weights<<<(DIN * HID + TPB - 1) / TPB, TPB, 0, stream>>>(W1, W2, w1t, w2t);
    edge_mlp_scatter<<<E / BE, TPB, 0, stream>>>(h_dk, h_si, h_sk, h_di, fidx, cidx,
                                                 w1t, w2t, b1, b2, out);
}
